// CaptioningRNN_86818468921881
// MI455X (gfx1250) — compile-verified, matrix operand order checked by reading, not run on hardware
//
#include <hip/hip_runtime.h>
#include <hip/hip_bf16.h>
#include <math.h>

// MI455X / gfx1250: wave32, WMMA 16x16x32 bf16 -> f32 accumulate.
typedef __bf16 bf16;
typedef __attribute__((ext_vector_type(16))) __bf16 v16bf;
typedef __attribute__((ext_vector_type(8)))  float  v8f;

#define N_B   128     // batch
#define T_ALL 33      // caption length
#define T_S   32      // LSTM steps (T-1)
#define D_F   1280    // feature dim
#define W_D   256     // embed dim
#define H_D   512     // hidden dim
#define V_D   32000   // vocab
#define FOURH 2048    // 4*H
#define KCAT  768     // W_D + H_D
#define NT_V  (V_D / 16)   // 2000 vocab column tiles
#define KC_V  (H_D / 32)   // 16 k-chunks for vocab GEMM
#define KC_L  (KCAT / 32)  // 24 k-chunks for LSTM gate GEMM

// Fragment-order packing: frag[lane][j] with
//   col  = tile*16 + (lane&15)
//   koff = (lane&16) ? 8 : 0
//   k    = kchunk*32 + koff + j + (j>=8 ? 8 : 0)
// so a per-lane v16bf load is one contiguous, 32B-aligned vector.

// ---- A fragment from row-major source (rows contiguous -> b128 loads) ----
__device__ inline v16bf load_a_frag(const bf16* __restrict__ src, int row, int ld,
                                    int K0, int lane) {
  int koff = (lane & 16) ? 8 : 0;
  const bf16* p = src + (size_t)row * ld + K0 + koff;
  v16bf a;
#pragma unroll
  for (int i = 0; i < 8; ++i) { a[i] = p[i]; a[i + 8] = p[16 + i]; }
  return a;
}

// online-softmax cross-lane merge over the 16 lanes sharing the same 8 rows
__device__ inline void merge16(float* m, float* s) {
#pragma unroll
  for (int mask = 1; mask < 16; mask <<= 1) {
#pragma unroll
    for (int r = 0; r < 8; ++r) {
      float om = __shfl_xor(m[r], mask, 32);
      float os = __shfl_xor(s[r], mask, 32);
      float M = fmaxf(m[r], om);
      float S = s[r] * __expf(m[r] - M) + os * __expf(om - M);
      m[r] = M; s[r] = S;
    }
  }
}

// ---- one-time weight conversion + fragment packing ----
__global__ void k_pack_weights(const float* __restrict__ Wx, const float* __restrict__ Wh,
                               const float* __restrict__ Wv,
                               bf16* __restrict__ Wcat_pk, bf16* __restrict__ Wv_pk) {
  size_t i0 = blockIdx.x * (size_t)blockDim.x + threadIdx.x;
  size_t stride = (size_t)gridDim.x * blockDim.x;
  // Wcat_pk: [nt=128][kci=24][lane=32][j=16]
  size_t ncat = (size_t)128 * KC_L * 32 * 16;
  for (size_t i = i0; i < ncat; i += stride) {
    int j = (int)(i & 15);
    int lane = (int)((i >> 4) & 31);
    size_t rest = i >> 9;
    int kci = (int)(rest % KC_L);
    int nt  = (int)(rest / KC_L);
    int col  = nt * 16 + (lane & 15);
    int koff = (lane & 16) ? 8 : 0;
    int k    = kci * 32 + koff + j + ((j >= 8) ? 8 : 0);
    float v = (k < W_D) ? Wx[(size_t)k * FOURH + col]
                        : Wh[(size_t)(k - W_D) * FOURH + col];
    Wcat_pk[i] = (bf16)v;
  }
  // Wv_pk: [nt=2000][kci=16][lane=32][j=16]
  size_t nvp = (size_t)NT_V * KC_V * 32 * 16;
  for (size_t i = i0; i < nvp; i += stride) {
    int j = (int)(i & 15);
    int lane = (int)((i >> 4) & 31);
    size_t rest = i >> 9;
    int kci = (int)(rest & (KC_V - 1));
    int nt  = (int)(rest >> 4);
    int col  = nt * 16 + (lane & 15);
    int koff = (lane & 16) ? 8 : 0;
    int k    = kci * 32 + koff + j + ((j >= 8) ? 8 : 0);
    Wv_pk[i] = (bf16)Wv[(size_t)k * V_D + col];
  }
}

// ---- embedding gather -> bf16, layout [t][n][k] ----
__global__ void k_embed(const int* __restrict__ captions, const float* __restrict__ Wemb,
                        bf16* __restrict__ xall) {
  size_t i = blockIdx.x * (size_t)blockDim.x + threadIdx.x;
  if (i >= (size_t)T_S * N_B * W_D) return;
  int k = (int)(i % W_D);
  size_t r = i / W_D;
  int n = (int)(r % N_B);
  int t = (int)(r / N_B);
  int tok = captions[n * T_ALL + t];
  xall[i] = (bf16)Wemb[(size_t)tok * W_D + k];
}

// ---- h0 = features @ W_proj + b_proj (tiny, f32), c0 = 0 ----
__global__ void k_h0(const float* __restrict__ feats, const float* __restrict__ Wp,
                     const float* __restrict__ bp, bf16* __restrict__ hbf,
                     float* __restrict__ cbuf) {
  int i = blockIdx.x * blockDim.x + threadIdx.x;
  if (i >= N_B * H_D) return;
  int n = i / H_D, j = i % H_D;
  float acc = bp[j];
  for (int k = 0; k < D_F; ++k) acc += feats[n * D_F + k] * Wp[k * H_D + j];
  hbf[i] = (bf16)acc;
  cbuf[i] = 0.0f;
}

// ---- per-step gate GEMM: Ag(128x2048) = [xt|h](128x768) @ Wcat(768x2048) + b ----
__global__ void __launch_bounds__(256)
k_lstm_gemm(const bf16* __restrict__ xall_t, const bf16* __restrict__ hbf,
            const bf16* __restrict__ Wcat_pk, const float* __restrict__ bias,
            float* __restrict__ Ag) {
  int lane = threadIdx.x & 31;
  int wave = threadIdx.x >> 5;
  int tile = blockIdx.x * 8 + wave;       // 1024 tiles = 8(M) x 128(N)
  int mt = tile >> 7, nt = tile & 127;
  int R0 = mt * 16, C0 = nt * 16;
  int col  = C0 + (lane & 15);
  int arow = R0 + (lane & 15);
  const bf16* bbase = Wcat_pk + (((size_t)nt * KC_L) * 32 + lane) * 16;
  v8f acc = {};
#pragma unroll
  for (int kci = 0; kci < KC_L; ++kci) {
    int kc = kci * 32;
    v16bf a = (kc < W_D) ? load_a_frag(xall_t, arow, W_D, kc, lane)
                         : load_a_frag(hbf,    arow, H_D, kc - W_D, lane);
    v16bf b = *(const v16bf*)(bbase + (size_t)kci * 32 * 16);
    acc = __builtin_amdgcn_wmma_f32_16x16x32_bf16(false, a, false, b,
                                                  (short)0, acc, false, false);
  }
  float bv = bias[col];
  int rbase = R0 + ((lane & 16) ? 8 : 0);
#pragma unroll
  for (int r = 0; r < 8; ++r)
    Ag[(size_t)(rbase + r) * FOURH + col] = acc[r] + bv;
}

// ---- gate nonlinearity + state update; writes h in row-major AND A-fragment order ----
__global__ void k_lstm_gates(const float* __restrict__ Ag, float* __restrict__ cbuf,
                             bf16* __restrict__ hbf, bf16* __restrict__ hs_t,
                             bf16* __restrict__ Apack, int t) {
  int i = blockIdx.x * blockDim.x + threadIdx.x;
  if (i >= N_B * H_D) return;
  int n = i / H_D, j = i % H_D;
  const float* a = Ag + (size_t)n * FOURH;
  float ig = 1.f / (1.f + __expf(-a[j]));
  float fg = 1.f / (1.f + __expf(-a[H_D + j]));
  float og = 1.f / (1.f + __expf(-a[2 * H_D + j]));
  float gg = tanhf(a[3 * H_D + j]);
  float c = fg * cbuf[i] + ig * gg;
  float h = og * tanhf(c);
  cbuf[i] = c;
  bf16 hb = (bf16)h;
  hbf[i]  = hb;   // next step's GEMM (row-major)
  hs_t[i] = hb;   // target-score dot (row-major)
  // A-fragment order for the vocab GEMM: row r = t*128+n, K index = j
  int r = t * N_B + n;
  int mtile = r >> 4, rr = r & 15;
  int kci = j >> 5, kk32 = j & 31;
  int half = (kk32 >> 3) & 1;
  int lane = rr + (half << 4);
  int koff = half << 3;
  int jj = (kk32 < 16) ? (kk32 - koff) : (kk32 - 8 - koff);
  Apack[(((size_t)mtile * KC_V + kci) * 32 + lane) * 16 + jj] = hb;
}

// ---- fused vocab GEMM + online log-softmax + target NLL ----
// One block per 32-row tile: TWO register-resident A fragment sets (256 VGPRs/lane),
// so each packed-B vector load feeds two WMMAs (halves L2 B traffic).
__global__ void __launch_bounds__(256)
k_vocab_loss(const bf16* __restrict__ Apack, const bf16* __restrict__ Wv_pk,
             const bf16* __restrict__ hsbf, const float* __restrict__ Wv,
             const float* __restrict__ bvoc, const int* __restrict__ captions,
             float* __restrict__ nll) {
  __shared__ float red_m[8][32];
  __shared__ float red_s[8][32];
  __shared__ float lse_s[32];

  int lane = threadIdx.x & 31;
  int wave = threadIdx.x >> 5;
  int r0 = blockIdx.x * 32;               // 128 blocks; row = t*128 + n
  int t  = r0 / N_B;
  int n0 = r0 % N_B;

  // Preload both 16-row tiles' A fragments into registers.
  v16bf a0[KC_V], a1[KC_V];
  const bf16* ab0 = Apack + ((size_t)(2 * blockIdx.x)     * KC_V * 32 + lane) * 16;
  const bf16* ab1 = Apack + ((size_t)(2 * blockIdx.x + 1) * KC_V * 32 + lane) * 16;
#pragma unroll
  for (int kci = 0; kci < KC_V; ++kci) {
    a0[kci] = *(const v16bf*)(ab0 + (size_t)kci * 512);
    a1[kci] = *(const v16bf*)(ab1 + (size_t)kci * 512);
  }

  int colLocal = lane & 15;
  float m0[8], s0[8], m1[8], s1[8];
#pragma unroll
  for (int r = 0; r < 8; ++r) {
    m0[r] = -INFINITY; s0[r] = 0.f;
    m1[r] = -INFINITY; s1[r] = 0.f;
  }

  for (int ntile = wave; ntile < NT_V; ntile += 8) {
    const bf16* bbase = Wv_pk + ((size_t)ntile * KC_V * 32 + lane) * 16;
    if (ntile + 8 < NT_V)                 // prefetch next B tile (global_prefetch_b8)
      __builtin_prefetch(bbase + (size_t)8 * KC_V * 512, 0, 1);
    v8f acc0 = {}, acc1 = {};
#pragma unroll
    for (int kci = 0; kci < KC_V; ++kci) {
      v16bf b = *(const v16bf*)(bbase + (size_t)kci * 512);
      acc0 = __builtin_amdgcn_wmma_f32_16x16x32_bf16(false, a0[kci], false, b,
                                                     (short)0, acc0, false, false);
      acc1 = __builtin_amdgcn_wmma_f32_16x16x32_bf16(false, a1[kci], false, b,
                                                     (short)0, acc1, false, false);
    }
    float bv = bvoc[ntile * 16 + colLocal];
#pragma unroll
    for (int r = 0; r < 8; ++r) {         // online softmax updates
      float v = acc0[r] + bv;
      if (v > m0[r]) { s0[r] = s0[r] * __expf(m0[r] - v) + 1.f; m0[r] = v; }
      else           { s0[r] += __expf(v - m0[r]); }
      float w = acc1[r] + bv;
      if (w > m1[r]) { s1[r] = s1[r] * __expf(m1[r] - w) + 1.f; m1[r] = w; }
      else           { s1[r] += __expf(w - m1[r]); }
    }
  }
  merge16(m0, s0);
  merge16(m1, s1);
  if (colLocal == 0) {
    int rbase = (lane & 16) ? 8 : 0;
#pragma unroll
    for (int r = 0; r < 8; ++r) {
      red_m[wave][rbase + r]      = m0[r];  red_s[wave][rbase + r]      = s0[r];
      red_m[wave][16 + rbase + r] = m1[r];  red_s[wave][16 + rbase + r] = s1[r];
    }
  }
  __syncthreads();
  if (threadIdx.x < 32) {                 // merge the 8 waves -> logsumexp per row
    float M = -INFINITY;
    for (int w = 0; w < 8; ++w) M = fmaxf(M, red_m[w][threadIdx.x]);
    float S = 0.f;
    for (int w = 0; w < 8; ++w) S += red_s[w][threadIdx.x] * __expf(red_m[w][threadIdx.x] - M);
    lse_s[threadIdx.x] = M + __logf(S);
  }
  __syncthreads();
  // target score: wave w computes rows 4w..4w+3 (bf16-rounded weights for consistency)
  for (int rr = wave * 4; rr < wave * 4 + 4; ++rr) {
    int n = n0 + rr;
    int tok = captions[n * T_ALL + (t + 1)];   // captions_out
    const bf16* hrow = hsbf + ((size_t)t * N_B + n) * H_D;
    float part = 0.f;
    for (int k = lane; k < H_D; k += 32)
      part += (float)hrow[k] * (float)(bf16)Wv[(size_t)k * V_D + tok];
#pragma unroll
    for (int mask = 16; mask >= 1; mask >>= 1) part += __shfl_xor(part, mask, 32);
    if (lane == 0) {
      float score = part + bvoc[tok];
      nll[r0 + rr] = (tok != 0) ? (lse_s[rr] - score) : 0.f;
    }
  }
}

// ---- deterministic single-block final reduction: loss = sum(nll)/N ----
__global__ void k_reduce(const float* __restrict__ nll, float* __restrict__ out) {
  __shared__ float sm[256];
  float acc = 0.f;
  for (int i = threadIdx.x; i < T_S * N_B; i += 256) acc += nll[i];
  sm[threadIdx.x] = acc;
  __syncthreads();
  for (int w = 128; w > 0; w >>= 1) {
    if ((int)threadIdx.x < w) sm[threadIdx.x] += sm[threadIdx.x + w];
    __syncthreads();
  }
  if (threadIdx.x == 0) out[0] = sm[0] / (float)N_B;
}

extern "C" void kernel_launch(void* const* d_in, const int* in_sizes, int n_in,
                              void* d_out, int out_size, void* d_ws, size_t ws_size,
                              hipStream_t stream) {
  const float* features = (const float*)d_in[0];
  const int*   captions = (const int*)d_in[1];
  const float* W_proj   = (const float*)d_in[2];
  const float* b_proj   = (const float*)d_in[3];
  const float* W_embed  = (const float*)d_in[4];
  const float* Wx       = (const float*)d_in[5];
  const float* Wh       = (const float*)d_in[6];
  const float* b        = (const float*)d_in[7];
  const float* W_vocab  = (const float*)d_in[8];
  const float* b_vocab  = (const float*)d_in[9];
  float* out = (float*)d_out;
  (void)in_sizes; (void)n_in; (void)out_size; (void)ws_size;

  char* ws = (char*)d_ws;
  size_t off = 0;
  auto alloc = [&](size_t bytes) -> void* {
    void* p = ws + off;
    off = (off + bytes + 255) & ~(size_t)255;
    return p;
  };
  bf16*  Wcat_pk = (bf16*) alloc((size_t)128 * KC_L * 32 * 16 * 2);  //  3.1 MB
  bf16*  Wv_pk   = (bf16*) alloc((size_t)NT_V * KC_V * 32 * 16 * 2); // 32.8 MB
  bf16*  xall    = (bf16*) alloc((size_t)T_S * N_B * W_D * 2);       //  2.1 MB
  bf16*  hbf     = (bf16*) alloc((size_t)N_B * H_D * 2);
  float* cbuf    = (float*)alloc((size_t)N_B * H_D * 4);
  bf16*  hsbf    = (bf16*) alloc((size_t)T_S * N_B * H_D * 2);       //  4.2 MB
  bf16*  Apack   = (bf16*) alloc((size_t)T_S * N_B * H_D * 2);       //  4.2 MB
  float* Ag      = (float*)alloc((size_t)N_B * FOURH * 4);           //  1.0 MB
  float* nll     = (float*)alloc((size_t)T_S * N_B * 4);

  k_pack_weights<<<2048, 256, 0, stream>>>(Wx, Wh, W_vocab, Wcat_pk, Wv_pk);
  k_embed<<<(T_S * N_B * W_D + 255) / 256, 256, 0, stream>>>(captions, W_embed, xall);
  k_h0<<<(N_B * H_D + 255) / 256, 256, 0, stream>>>(features, W_proj, b_proj, hbf, cbuf);
  for (int t = 0; t < T_S; ++t) {
    k_lstm_gemm<<<128, 256, 0, stream>>>(xall + (size_t)t * N_B * W_D, hbf, Wcat_pk, b, Ag);
    k_lstm_gates<<<(N_B * H_D + 255) / 256, 256, 0, stream>>>(
        Ag, cbuf, hbf, hsbf + (size_t)t * N_B * H_D, Apack, t);
  }
  k_vocab_loss<<<128, 256, 0, stream>>>(Apack, Wv_pk, hsbf, W_vocab, b_vocab, captions, nll);
  k_reduce<<<1, 256, 0, stream>>>(nll, out);
}
